// GRANDE_83013127897543
// MI455X (gfx1250) — compile-verified, matrix operand order checked by reading, not run on hardware
//
#include <hip/hip_runtime.h>

// GRANDE forward, collapsed to its straight-through-hardened form:
//   v*[e,i]   = argmax_v logits[e,i,v]            (entmax15 is monotone -> same argmax)
//   s1[e,i]   = split_values[e,i,v*]
//   feat[e,i] = features_by_estimator[e, v*]
//   per (b,e): walk depth-6 heap, bit = (s1 <= x[feat]), leaf = 6 bits
//   w[b,e]    = estimator_weights[e, leaf];  p = softmax_e(w)
//   out[b,c]  = sum_e p[b,e] * leaf_classes[e, leaf(b,e), c]
// Final 256-term reduction is done with chained v_wmma_f32_16x16x32_f16
// (all-ones A => every D row holds the column sums; layout-permutation safe).

#define B_  2048
#define E_  256
#define DEPTH_ 6
#define V_  50
#define F_  200
#define C_  10
#define I_  63   // 2^DEPTH - 1
#define L_  64   // 2^DEPTH

typedef __attribute__((ext_vector_type(16))) _Float16 v16h;
typedef __attribute__((ext_vector_type(8)))  float    v8f;

// ---------------- Kernel 1: per (e,i) hard split selection -----------------
__global__ __launch_bounds__(256)
void grande_select_kernel(const float* __restrict__ logits,       // (E,I,V)
                          const float* __restrict__ split_values, // (E,I,V)
                          const int*   __restrict__ fbe,          // (E,V)
                          float* __restrict__ s1f,                // (E*I)
                          int*   __restrict__ feat) {             // (E*I)
  int t = blockIdx.x * blockDim.x + threadIdx.x;
  if (t >= E_ * I_) return;
  const float* row = logits + (size_t)t * V_;
  float best = row[0];
  int   bi   = 0;
  for (int v = 1; v < V_; ++v) {          // strict '>' keeps first max (jnp.argmax)
    float x = row[v];
    if (x > best) { best = x; bi = v; }
  }
  s1f[t]  = split_values[(size_t)t * V_ + bi];
  int e   = t / I_;
  feat[t] = fbe[e * V_ + bi];
}

// --------- Kernel 2: tree walk + softmax + WMMA class reduction ------------
// One wave32 per sample b; 8 waves (256 threads) per block.
__global__ __launch_bounds__(256)
void grande_main_kernel(const float* __restrict__ inputs, // (B,F)
                        const float* __restrict__ s1f,    // (E*I)
                        const int*   __restrict__ feat,   // (E*I)
                        const float* __restrict__ ew,     // (E,L)
                        const float* __restrict__ lc,     // (E,L,C)
                        float* __restrict__ out) {        // (B,C)
  __shared__ float pbuf[8][E_];
  __shared__ int   lbuf[8][E_];

  const int lane = threadIdx.x & 31;
  const int wv   = threadIdx.x >> 5;
  const int b    = blockIdx.x * 8 + wv;
  const float* xrow = inputs + (size_t)b * F_;
  __builtin_prefetch(xrow, 0, 3);   // global_prefetch_b8: feature row is hot

  // --- 8 tree traversals per lane (e = lane + 32*j) ---
  float wl[8];
  int   lf[8];
  #pragma unroll
  for (int j = 0; j < 8; ++j) {
    const int e    = lane + 32 * j;
    const int base = e * I_;
    int k = 0;
    #pragma unroll
    for (int lvl = 0; lvl < DEPTH_; ++lvl) {
      const int idx = base + (1 << lvl) - 1 + k;
      const float s = s1f[idx];
      const float x = xrow[feat[idx]];
      k = 2 * k + ((s <= x) ? 1 : 0);   // leaf bit = 1 - node, tie -> 1-0 path
    }
    lf[j] = k;
    wl[j] = ew[e * L_ + k];
  }

  // --- wave32 softmax over the 256 estimator weights of this sample ---
  float m = wl[0];
  #pragma unroll
  for (int j = 1; j < 8; ++j) m = fmaxf(m, wl[j]);
  #pragma unroll
  for (int off = 16; off >= 1; off >>= 1) m = fmaxf(m, __shfl_xor(m, off, 32));

  float pe[8];
  float s = 0.f;
  #pragma unroll
  for (int j = 0; j < 8; ++j) { pe[j] = __expf(wl[j] - m); s += pe[j]; }
  #pragma unroll
  for (int off = 16; off >= 1; off >>= 1) s += __shfl_xor(s, off, 32);
  const float inv = 1.0f / s;

  #pragma unroll
  for (int j = 0; j < 8; ++j) {
    const int e = lane + 32 * j;
    pbuf[wv][e] = pe[j] * inv;
    lbuf[wv][e] = lf[j];
  }
  __syncthreads();   // make per-wave LDS region cross-lane visible

  // --- WMMA reduction: out[b,c] = sum_e p[e] * LC[e, leaf_e, c] ---
  // A = ones(16x32) f16; B tile = 32 e-rows x 16 classes (10 real + 6 zero).
  // With all-ones A, D[m,n] = sum_k B[k,n] for every row m.
  v16h a;
  #pragma unroll
  for (int i = 0; i < 16; ++i) a[i] = (_Float16)1.0f;

  v8f acc = {};
  const int c     = lane & 15;
  const int cc    = (c < C_) ? c : 0;          // clamp to stay in-bounds
  const int khalf = (lane & 16) ? 16 : 0;      // lanes 16-31 carry K=16..31

  #pragma unroll
  for (int t = 0; t < 8; ++t) {
    v16h bm;
    #pragma unroll
    for (int kk = 0; kk < 16; ++kk) {
      const int e   = t * 32 + khalf + kk;
      float val = pbuf[wv][e] * lc[(size_t)(e * L_ + lbuf[wv][e]) * C_ + cc];
      if (c >= C_) val = 0.f;
      bm[kk] = (_Float16)val;
    }
    // (neg_a, A, neg_b, B, c_mod, C, reuse_a, reuse_b)
    acc = __builtin_amdgcn_wmma_f32_16x16x32_f16(
        false, a, false, bm, (short)0, acc, false, false);
  }

  // Every D row is identical; D[0, c] lives in acc[0] with N = lane%16.
  if (lane < C_) out[b * C_ + lane] = acc[0];
}

// ---------------------------------------------------------------------------
extern "C" void kernel_launch(void* const* d_in, const int* in_sizes, int n_in,
                              void* d_out, int out_size, void* d_ws, size_t ws_size,
                              hipStream_t stream) {
  const float* inputs = (const float*)d_in[0];  // (B,F)
  const int*   fbe    = (const int*)  d_in[1];  // (E,V)
  const float* sv     = (const float*)d_in[2];  // (E,I,V)
  const float* logits = (const float*)d_in[3];  // (E,I,V)
  const float* ew     = (const float*)d_in[4];  // (E,L)
  const float* lc     = (const float*)d_in[5];  // (E,L,C)
  // d_in[6] path_identifier / d_in[7] internal_node_index: fixed heap layout,
  // derived analytically in the kernel.
  float* out = (float*)d_out;

  float* s1f  = (float*)d_ws;
  int*   feat = (int*)((char*)d_ws + sizeof(float) * (size_t)(E_ * I_));

  const int n1 = E_ * I_;
  grande_select_kernel<<<(n1 + 255) / 256, 256, 0, stream>>>(logits, sv, fbe, s1f, feat);
  grande_main_kernel<<<B_ / 8, 256, 0, stream>>>(inputs, s1f, feat, ew, lc, out);

  (void)in_sizes; (void)n_in; (void)out_size; (void)ws_size;
}
